// LorentzPerHeadAttention_78829829750903
// MI455X (gfx1250) — matrix operand order repaired and checked by reading, not compiled, not run on hardware
//
#include <hip/hip_runtime.h>

// ---------------------------------------------------------------------------
// Lorentz per-head attention for MI455X (gfx1250, wave32, WMMA).
//   B=2, S=2048, D_MODEL=512, H=8, DH=64.  ~23 GFLOP, ~50MB working set
//   (fully L2-resident at 192MB) -> matrix-op bound; all GEMM-shaped work
//   runs on v_wmma_f32_16x16x32_f16 (f16 in, f32 accumulate), and all tile
//   staging uses the CDNA5 async global->LDS engine (ASYNCcnt) when the
//   toolchain exposes it.
// Score trick: score = (c/8)*(Qt*Kt - Qs.Ks) folded into a 96-dim padded
//   contraction qhat.khat, so QK^T is pure WMMA (3 K-chunks of 32).
// ---------------------------------------------------------------------------

#define S_LEN  2048
#define DMODEL 512
#define NH     8
#define DH     64
#define DQK    96          // 64 spatial + 1 time + 31 zero pad (3 x K=32)
#define NB     2
#define NEGBIG (-1.0e30f)

typedef __attribute__((ext_vector_type(16))) _Float16 v16h;
typedef __attribute__((ext_vector_type(8)))  float    v8f;
typedef __attribute__((__vector_size__(16))) int      asy_i4;   // builtin's pointee type

union F16Frag { v16h v; uint4 q[2]; _Float16 h[16]; };

__device__ __forceinline__ v16h load_frag_2x16B(const _Float16* p0, const _Float16* p1) {
    F16Frag f;
    f.q[0] = *(const uint4*)p0;
    f.q[1] = *(const uint4*)p1;
    return f.v;
}

__device__ __forceinline__ v8f wmma32(v16h a, v16h b, v8f c) {
    return __builtin_amdgcn_wmma_f32_16x16x32_f16(
        /*neg_a=*/false, a, /*neg_b=*/false, b,
        /*c_mod=*/(short)0, c, /*reuse_a=*/false, /*reuse_b=*/false);
}

// ---- CDNA5 async global->LDS staging (ASYNCcnt path), with sync fallback ---
#if defined(__HIP_DEVICE_COMPILE__) && \
    __has_builtin(__builtin_amdgcn_global_load_async_to_lds_b128) && \
    __has_builtin(__builtin_amdgcn_s_wait_asynccnt)
#define ASYNC_LDS_OK 1
#endif

__device__ __forceinline__ void stage_b128(const void* g, void* l) {
#ifdef ASYNC_LDS_OK
    __builtin_amdgcn_global_load_async_to_lds_b128(
        (__attribute__((address_space(1))) asy_i4*)(void*)g,
        (__attribute__((address_space(3))) asy_i4*)l, 0, 0);
#else
    *(uint4*)l = *(const uint4*)g;
#endif
}

__device__ __forceinline__ void stage_wait() {
#ifdef ASYNC_LDS_OK
    __builtin_amdgcn_s_wait_asynccnt(0);
#endif
}

// ---------------------------------------------------------------------------
// K1: Lorentz log map at origin (ambient K=-1): x (B,S,513) f32 -> x_eu f16.
// ---------------------------------------------------------------------------
__global__ __launch_bounds__(256) void logmap_kernel(const float* __restrict__ x,
                                                     _Float16* __restrict__ xeu) {
    const int row  = blockIdx.x;                 // 0..B*S-1
    const int tid  = threadIdx.x;
    const int lane = tid & 31, w = tid >> 5;
    const float* xr = x + (size_t)row * (DMODEL + 1);
    const float  xt = xr[0];

    float ss = 0.f;
    for (int i = tid; i < DMODEL; i += 256) { float v = xr[1 + i]; ss += v * v; }
    #pragma unroll
    for (int off = 16; off >= 1; off >>= 1) ss += __shfl_xor(ss, off, 32);

    __shared__ float red[8];
    __shared__ float s_scale;
    if (lane == 0) red[w] = ss;
    __syncthreads();
    if (tid == 0) {
        float t = 0.f;
        #pragma unroll
        for (int i = 0; i < 8; i++) t += red[i];
        float nrm   = fmaxf(sqrtf(t), 1e-7f);
        float theta = acoshf(fmaxf(xt, 1.0f + 1e-7f));   // sc = 1 (K=-1)
        s_scale = theta / nrm;
    }
    __syncthreads();
    const float k = s_scale;
    for (int i = tid; i < DMODEL; i += 256)
        xeu[(size_t)row * DMODEL + i] = (_Float16)(k * xr[1 + i]);
}

// ---------------------------------------------------------------------------
// K2: transpose f32 weight (512x512) -> f16 Wt[n*512+k] = W[k*512+n].
// ---------------------------------------------------------------------------
__global__ __launch_bounds__(256) void wtrans_kernel(const float* __restrict__ W,
                                                     _Float16* __restrict__ Wt) {
    int idx = blockIdx.x * 256 + threadIdx.x;    // over 512*512
    int n = idx >> 9, k = idx & 511;
    Wt[idx] = (_Float16)W[(size_t)k * DMODEL + n];
}

// ---------------------------------------------------------------------------
// K3: GEMM  C_f32[M,N] = A_f16[M,Kd] @ Bt_f16[N,Kd]^T  (Bt stored row=N,col=K)
// Block tile 128x128, 8 waves -> each wave a 32x64 C patch (8 WMMA / K-chunk
// from 2 A-frags x 4 B-frags). Tiles staged with async b128 copies.
// ---------------------------------------------------------------------------
__global__ __launch_bounds__(256) void gemm_f16_f32_kernel(
        const _Float16* __restrict__ A, const _Float16* __restrict__ Bt,
        float* __restrict__ C, int M, int N, int Kd) {
    __shared__ _Float16 lA[128 * 32];            // 8 KB
    __shared__ _Float16 lB[128 * 32];            // 8 KB
    const int tid = threadIdx.x, lane = tid & 31, w = tid >> 5;
    const int ln = lane & 15, hi = lane >> 4;
    const int m0 = blockIdx.x * 128, n0 = blockIdx.y * 128;
    const int mi = w & 3, nj = w >> 2;

    v8f acc[8] = {{}, {}, {}, {}, {}, {}, {}, {}};

    for (int kk = 0; kk < Kd; kk += 32) {
        #pragma unroll
        for (int i = 0; i < 2; i++) {            // 512 x 16B per tile
            int idx = tid + i * 256;
            int r = idx >> 2, c4 = idx & 3;
            stage_b128(A  + (size_t)(m0 + r) * Kd + kk + c4 * 8, lA + idx * 8);
            stage_b128(Bt + (size_t)(n0 + r) * Kd + kk + c4 * 8, lB + idx * 8);
        }
        stage_wait();
        __syncthreads();

        const _Float16* ap0 = lA + (mi * 32 + 0  + ln) * 32 + hi * 8;
        const _Float16* ap1 = lA + (mi * 32 + 16 + ln) * 32 + hi * 8;
        v16h a0 = load_frag_2x16B(ap0, ap0 + 16);
        v16h a1 = load_frag_2x16B(ap1, ap1 + 16);
        #pragma unroll
        for (int j = 0; j < 4; j++) {
            const _Float16* bp = lB + (nj * 64 + j * 16 + ln) * 32 + hi * 8;
            v16h bf = load_frag_2x16B(bp, bp + 16);
            acc[j]     = wmma32(a0, bf, acc[j]);
            acc[4 + j] = wmma32(a1, bf, acc[4 + j]);
        }
        __syncthreads();
    }
    #pragma unroll
    for (int t16 = 0; t16 < 2; t16++)
        #pragma unroll
        for (int j = 0; j < 4; j++)
            #pragma unroll
            for (int r = 0; r < 8; r++) {
                int row = m0 + mi * 32 + t16 * 16 + r + 8 * hi;
                int col = n0 + nj * 64 + j * 16 + ln;
                C[(size_t)row * N + col] = acc[t16 * 4 + j][r];
            }
}

// ---------------------------------------------------------------------------
// K4: per-head Lorentz exp map at origin + score folding.
// In:  U f32 laid out (b,s,h,d). Out f16 laid out (b,h,s,96):
//   isQ: [-(c/8)*sinh(sc n)/(sc n)*u , (c/8)*cosh(sc n)/sc , 0...]
//   !isQ:[  sinh(sc n)/(sc n)*u      ,  cosh(sc n)/sc      , 0...]
// ---------------------------------------------------------------------------
__global__ __launch_bounds__(32) void expmap_kernel(const float* __restrict__ U,
                                                    const float* __restrict__ logK,
                                                    _Float16* __restrict__ Out,
                                                    int isQ) {
    const int rid = blockIdx.x;                  // ((b*S+s)*H + h)
    const int t   = threadIdx.x;
    const int h   = rid & 7;
    const int s   = (rid >> 3) & (S_LEN - 1);
    const int b   = rid >> 14;                   // S*H = 16384
    const float* u = U + (size_t)rid * DH;
    const float u0 = u[t], u1 = u[t + 32];

    float ss = u0 * u0 + u1 * u1;
    #pragma unroll
    for (int off = 16; off >= 1; off >>= 1) ss += __shfl_xor(ss, off, 32);

    const float c   = __expf(logK[h]);           // abs(K_h)
    const float sc  = sqrtf(c);
    const float n   = fmaxf(sqrtf(ss), 1e-7f);
    const float scn = sc * n;
    const float tc  = coshf(scn) / sc;
    const float sf  = sinhf(scn) / scn;
    const float qs  = isQ ? (c * 0.125f) : 1.0f; // fold abs_K / sqrt(64)
    const float sg  = isQ ? -1.0f : 1.0f;

    _Float16* o = Out + ((size_t)((b << 3) + h) * S_LEN + s) * DQK;
    o[t]      = (_Float16)(sg * qs * sf * u0);
    o[t + 32] = (_Float16)(sg * qs * sf * u1);
    o[64 + t] = (t == 0) ? (_Float16)(qs * tc) : (_Float16)0.0f;  // pad 65..95 = 0
}

// ---------------------------------------------------------------------------
// K5: V f32 (b,s,h,d) -> f16 tiled-transposed (b,h, s/32, d, 32), so each
// 32-key attention tile is one contiguous 4KB [d][k] block (async-stageable).
// ---------------------------------------------------------------------------
__global__ __launch_bounds__(256) void vconv_kernel(const float* __restrict__ Vf,
                                                    _Float16* __restrict__ Vh) {
    int idx = blockIdx.x * 256 + threadIdx.x;    // over B*H*S*DH = 2^21
    int kk = idx & 31;
    int d  = (idx >> 5) & 63;
    int kt = (idx >> 11) & 63;                   // S/32 = 64
    int bh = idx >> 17;
    int b = bh >> 3, h = bh & 7;
    int s = kt * 32 + kk;
    Vh[idx] = (_Float16)Vf[((((size_t)b * S_LEN + s) * NH + h) << 6) + d];
}

// ---------------------------------------------------------------------------
// K6: flash causal attention. Grid (S/128, B*H); 256 thr = 8 waves.
// Each wave owns 16 query rows; Q fragments (16x96) live in VGPRs.
// 32-key tiles async-staged in LDS (Khat row-major; V pre-transposed tiles).
// Online softmax via half-wave shuffles; P relayout through per-wave LDS.
// ---------------------------------------------------------------------------
__global__ __launch_bounds__(256) void attn_kernel(const _Float16* __restrict__ Qh,
                                                   const _Float16* __restrict__ Kh,
                                                   const _Float16* __restrict__ Vh,
                                                   _Float16* __restrict__ Oh) {
    __shared__ _Float16 lK[32 * DQK];            // 6144 B, [key][96]
    __shared__ _Float16 lV[DH * 32];             // 4096 B, [d][key]  (transposed)
    __shared__ _Float16 lP[8][16 * 32];          // 8192 B, per-wave P staging

    const int tid = threadIdx.x, lane = tid & 31, w = tid >> 5;
    const int ln = lane & 15, hi = lane >> 4;
    const int qbase = blockIdx.x * 128;
    const int bh = blockIdx.y, b = bh >> 3, h = bh & 7;
    const int qrow = qbase + w * 16;

    const _Float16* Qb = Qh + (size_t)bh * S_LEN * DQK;
    const _Float16* Kb = Kh + (size_t)bh * S_LEN * DQK;
    const _Float16* Vb = Vh + (size_t)bh * S_LEN * DH;   // tiled-transposed

    // Q fragments: 16x96 = three 16x32 A-frags, loaded straight from global.
    v16h qa[3];
    {
        const _Float16* rp = Qb + (size_t)(qrow + ln) * DQK;
        #pragma unroll
        for (int c3 = 0; c3 < 3; c3++) {
            int base = c3 * 32 + hi * 8;
            qa[c3] = load_frag_2x16B(rp + base, rp + base + 16);
        }
    }

    v8f acc[4] = {{}, {}, {}, {}};
    float m_i[8], l_i[8];
    #pragma unroll
    for (int r = 0; r < 8; r++) { m_i[r] = NEGBIG; l_i[r] = 0.f; }

    const int nkt = (qbase + 128) >> 5;          // causal tile bound (block-level)
    for (int kt = 0; kt < nkt; ++kt) {
        const int k0 = kt * 32;
        // ---- async-stage K tile (6144B) and pre-transposed V tile (4096B) --
        const char* gK = (const char*)(Kb + (size_t)k0 * DQK);
        #pragma unroll
        for (int i = 0; i < 2; i++) {
            int idx = tid + i * 256;
            if (idx < 384) stage_b128(gK + idx * 16, (char*)lK + idx * 16);
        }
        const char* gV = (const char*)(Vb + (size_t)kt * DH * 32);
        stage_b128(gV + tid * 16, (char*)lV + tid * 16);
        stage_wait();
        __syncthreads();

        // ---- scores: 16 rows x 32 keys via 6 WMMAs ----
        v8f s0 = {}, s1 = {};
        #pragma unroll
        for (int c3 = 0; c3 < 3; c3++) {
            const _Float16* kp0 = lK + (size_t)ln * DQK + c3 * 32 + hi * 8;
            const _Float16* kp1 = lK + (size_t)(16 + ln) * DQK + c3 * 32 + hi * 8;
            s0 = wmma32(qa[c3], load_frag_2x16B(kp0, kp0 + 16), s0);
            s1 = wmma32(qa[c3], load_frag_2x16B(kp1, kp1 + 16), s1);
        }

        // ---- causal mask + online softmax (rows live in 16-lane halves) ----
        #pragma unroll
        for (int r = 0; r < 8; r++) {
            const int q_idx = qrow + r + 8 * hi;
            const int kA = k0 + ln, kB = k0 + 16 + ln;
            float v0 = (kA <= q_idx) ? s0[r] : NEGBIG;
            float v1 = (kB <= q_idx) ? s1[r] : NEGBIG;
            float mx = fmaxf(v0, v1);
            #pragma unroll
            for (int off = 1; off < 16; off <<= 1) mx = fmaxf(mx, __shfl_xor(mx, off, 16));
            const float mnew  = fmaxf(m_i[r], mx);
            const float alpha = __expf(m_i[r] - mnew);
            const float p0 = __expf(v0 - mnew);
            const float p1 = __expf(v1 - mnew);
            float rs = p0 + p1;
            #pragma unroll
            for (int off = 1; off < 16; off <<= 1) rs += __shfl_xor(rs, off, 16);
            l_i[r] = l_i[r] * alpha + rs;
            m_i[r] = mnew;
            const int m = r + 8 * hi;            // C layout row
            lP[w][m * 32 + ln]      = (_Float16)p0;
            lP[w][m * 32 + 16 + ln] = (_Float16)p1;
            #pragma unroll
            for (int g = 0; g < 4; g++) acc[g][r] *= alpha;
        }

        // ---- P (16x32) as A-frag, V chunks as B-frags: 4 WMMAs ----
        const _Float16* pp = &lP[w][ln * 32 + hi * 8];
        v16h pa = load_frag_2x16B(pp, pp + 16);
        #pragma unroll
        for (int g = 0; g < 4; g++) {
            const _Float16* vp = lV + (g * 16 + ln) * 32 + hi * 8;
            acc[g] = wmma32(pa, load_frag_2x16B(vp, vp + 16), acc[g]);
        }
        __syncthreads();
    }

    // ---- epilogue: normalize, write (b, s, h*64+d) f16 for final GEMM ----
    _Float16* Ob = Oh + (size_t)b * S_LEN * DMODEL + h * DH;
    #pragma unroll
    for (int g = 0; g < 4; g++)
        #pragma unroll
        for (int r = 0; r < 8; r++) {
            const int q = qrow + r + 8 * hi;
            Ob[(size_t)q * DMODEL + g * 16 + ln] = (_Float16)(acc[g][r] / l_i[r]);
        }
}

// ---------------------------------------------------------------------------
// Host side
// ---------------------------------------------------------------------------
extern "C" void kernel_launch(void* const* d_in, const int* in_sizes, int n_in,
                              void* d_out, int out_size, void* d_ws, size_t ws_size,
                              hipStream_t stream) {
    (void)in_sizes; (void)n_in; (void)out_size; (void)ws_size;
    const float* x    = (const float*)d_in[0];
    const float* Wq   = (const float*)d_in[1];
    const float* Wk   = (const float*)d_in[2];
    const float* Wv   = (const float*)d_in[3];
    const float* Wo   = (const float*)d_in[4];
    const float* logK = (const float*)d_in[5];

    const int ROWS = NB * S_LEN;                 // 4096
    char* ws = (char*)d_ws;
    size_t off = 0;
    auto alloc = [&](size_t bytes) { char* p = ws + off; off += (bytes + 255) & ~(size_t)255; return p; };

    _Float16* xeu_h = (_Float16*)alloc((size_t)ROWS * DMODEL * 2);        // 4 MB
    _Float16* WqT   = (_Float16*)alloc((size_t)DMODEL * DMODEL * 2);
    _Float16* WkT   = (_Float16*)alloc((size_t)DMODEL * DMODEL * 2);
    _Float16* WvT   = (_Float16*)alloc((size_t)DMODEL * DMODEL * 2);
    _Float16* WoT   = (_Float16*)alloc((size_t)DMODEL * DMODEL * 2);
    float*    Qf    = (float*)alloc((size_t)ROWS * DMODEL * 4);           // 8 MB
    float*    Kf    = (float*)alloc((size_t)ROWS * DMODEL * 4);
    float*    Vf    = (float*)alloc((size_t)ROWS * DMODEL * 4);
    _Float16* Qhat  = (_Float16*)alloc((size_t)NB * NH * S_LEN * DQK * 2); // 6 MB
    _Float16* Khat  = (_Float16*)alloc((size_t)NB * NH * S_LEN * DQK * 2);
    _Float16* Vh    = (_Float16*)alloc((size_t)NB * NH * S_LEN * DH * 2);  // 4 MB
    _Float16* attnO = (_Float16*)alloc((size_t)ROWS * DMODEL * 2);         // 4 MB

    // 1) log map -> f16 activations
    logmap_kernel<<<ROWS, 256, 0, stream>>>(x, xeu_h);

    // 2) weights -> transposed f16
    const int wgrid = (DMODEL * DMODEL) / 256;
    wtrans_kernel<<<wgrid, 256, 0, stream>>>(Wq, WqT);
    wtrans_kernel<<<wgrid, 256, 0, stream>>>(Wk, WkT);
    wtrans_kernel<<<wgrid, 256, 0, stream>>>(Wv, WvT);
    wtrans_kernel<<<wgrid, 256, 0, stream>>>(Wo, WoT);

    // 3) Q/K/V projections (4096x512x512 WMMA GEMMs)
    dim3 ggrid(ROWS / 128, DMODEL / 128);
    gemm_f16_f32_kernel<<<ggrid, 256, 0, stream>>>(xeu_h, WqT, Qf, ROWS, DMODEL, DMODEL);
    gemm_f16_f32_kernel<<<ggrid, 256, 0, stream>>>(xeu_h, WkT, Kf, ROWS, DMODEL, DMODEL);
    gemm_f16_f32_kernel<<<ggrid, 256, 0, stream>>>(xeu_h, WvT, Vf, ROWS, DMODEL, DMODEL);

    // 4) exp map + score folding -> 96-dim padded qhat/khat, (b,h,s,96)
    expmap_kernel<<<NB * S_LEN * NH, 32, 0, stream>>>(Qf, logK, Qhat, 1);
    expmap_kernel<<<NB * S_LEN * NH, 32, 0, stream>>>(Kf, logK, Khat, 0);

    // 5) V layout conversion -> tiled-transposed f16
    vconv_kernel<<<(NB * NH * S_LEN * DH) / 256, 256, 0, stream>>>(Vf, Vh);

    // 6) flash causal attention
    dim3 agrid(S_LEN / 128, NB * NH);
    attn_kernel<<<agrid, 256, 0, stream>>>(Qhat, Khat, Vh, attnO);

    // 7) output projection -> d_out (f32)
    gemm_f16_f32_kernel<<<ggrid, 256, 0, stream>>>(attnO, WoT, (float*)d_out,
                                                   ROWS, DMODEL, DMODEL);
}
